// duin_vqvae_68599217652118
// MI455X (gfx1250) — matrix operand order repaired and checked by reading, not compiled, not run on hardware
//
#include <hip/hip_runtime.h>
#include <hip/hip_bf16.h>
#include <math.h>

// ---------------- model dims ----------------
#define B_   32
#define SEQ_ 3000
#define C_   64
#define P_   100
#define L_   30           // SEQ/P tokens
#define D_   768
#define H_   12
#define DH_  64
#define FF_  3072
#define NL_  4
#define NS_  12
#define K_   8192
#define DC_  64
#define PC_  6400         // P*C
#define MTOK_ (B_*L_)     // 960 tokens
#define LN_EPS_ 1e-5f

typedef __attribute__((ext_vector_type(16))) __bf16 v16bf;
typedef __attribute__((ext_vector_type(8)))  __bf16 v8bf;
typedef __attribute__((ext_vector_type(4)))  __bf16 v4bf;
typedef __attribute__((ext_vector_type(2)))  __bf16 v2bf;
typedef __attribute__((ext_vector_type(8)))  float  v8f;

// epilogue flags
#define F_BIAS 1
#define F_RES  2
#define F_GELU 4
#define F_PE   8
#define F_SUBJ 16

static __device__ __forceinline__ float wave_sum(float v) {
  #pragma unroll
  for (int m = 16; m > 0; m >>= 1) v += __shfl_xor(v, m, 32);
  return v;
}
static __device__ __forceinline__ float wave_max(float v) {
  #pragma unroll
  for (int m = 16; m > 0; m >>= 1) v = fmaxf(v, __shfl_xor(v, m, 32));
  return v;
}

// =====================================================================
// Generic tiled GEMM: C[M,N] = epilogue(A[M,K] @ W[K,N])
// fp32 in/out, bf16 WMMA math (v_wmma_f32_16x16x32_bf16).
// Block = 128 threads (4 waves). Block tile = 16(M) x 128(N), K-step 32.
// Each wave owns two 16x16 N-subtiles -> 2 WMMAs / K-step.
// LDS: A row-major (lane fragment = two contiguous 16B runs -> ds_load_b128),
//      B transposed  (lane fragment = two contiguous 16B runs   -> ds_load_b128).
// Edge handling is branch-free: OOB rows/cols read CLAMPED (valid) addresses,
// producing garbage only in output rows/cols that are guarded at the store.
// F_SUBJ: grid.z = batch; A/C offset by z*M rows; W/bias gathered by subj[z].
// =====================================================================
__global__ __launch_bounds__(128)
void k_gemm(const float* __restrict__ A, int lda,
            const float* __restrict__ W, int ldw,
            const float* __restrict__ bias,
            const float* __restrict__ res,
            float* __restrict__ Cout, int ldc,
            int M, int N, int K, int flags,
            const int* __restrict__ subj, long wsub, int bsub)
{
  __shared__ __bf16 lA[16 * 32];    // [row][k]       row-major
  __shared__ __bf16 lBt[128 * 32];  // [col][k]       transposed

  const int tid  = threadIdx.x;
  const int lane = tid & 31;
  const int wave = tid >> 5;
  const int n0   = blockIdx.x * 128;
  const int m0   = blockIdx.y * 16;
  const int bz   = blockIdx.z;

  if (flags & F_SUBJ) {
    const int s = subj[bz];
    W    += (long)s * wsub;
    bias += (long)s * bsub;
    A    += (long)bz * M * lda;
    Cout += (long)bz * M * ldc;
    if (res) res += (long)bz * M * ldc;
  }

  v8f acc0 = {0.f, 0.f, 0.f, 0.f, 0.f, 0.f, 0.f, 0.f};
  v8f acc1 = {0.f, 0.f, 0.f, 0.f, 0.f, 0.f, 0.f, 0.f};

  // A staging coords (4 consecutive fp32 per thread, row clamped -> branch-free)
  const int a_r = (tid * 4) >> 5;
  const int a_c = (tid * 4) & 31;
  const int a_m = min(m0 + a_r, M - 1);
  const float* a_src = A + (long)a_m * lda + a_c;

  for (int k0 = 0; k0 < K; k0 += 32) {
    // ---- stage A tile (16x32): float4 load -> 4 bf16 -> ds_store_b64 ----
    {
      const float4 g = *(const float4*)(a_src + k0);
      v4bf p;
      p[0] = (__bf16)g.x; p[1] = (__bf16)g.y;
      p[2] = (__bf16)g.z; p[3] = (__bf16)g.w;
      *(v4bf*)&lA[a_r * 32 + a_c] = p;
    }
    // ---- stage B tile (32x128) transposed: 2x2 blocks, columns clamped ----
    #pragma unroll
    for (int i = 0; i < 8; ++i) {
      const int bb = tid + i * 128;          // 0..1023
      const int kr = bb >> 6;                // k pair 0..15
      const int nc = bb & 63;                // n pair 0..63
      const int nn = min(n0 + 2 * nc, N - 2);
      const float* wp = W + (long)(k0 + 2 * kr) * ldw + nn;
      const float2 g0 = *(const float2*)wp;
      const float2 g1 = *(const float2*)(wp + ldw);
      v2bf q0, q1;  // packed (k, k+1) for two adjacent columns
      q0[0] = (__bf16)g0.x; q0[1] = (__bf16)g1.x;   // col nn
      q1[0] = (__bf16)g0.y; q1[1] = (__bf16)g1.y;   // col nn+1
      *(v2bf*)&lBt[(2 * nc + 0) * 32 + 2 * kr] = q0;
      *(v2bf*)&lBt[(2 * nc + 1) * 32 + 2 * kr] = q1;
    }
    __syncthreads();

    if (k0 + 32 < K)  // speculative prefetch of next W tile -> global_prefetch_b8
      __builtin_prefetch(&W[(long)(k0 + 32 + (tid >> 2)) * ldw + n0 + ((tid & 3) << 5)], 0, 1);

    // ---- A fragment (ISA 16x32 layout): two contiguous 16B runs ----
    const int row = lane & 15;
    const int kb  = (lane < 16) ? 0 : 8;
    v8bf aLo = *(const v8bf*)&lA[row * 32 + kb];
    v8bf aHi = *(const v8bf*)&lA[row * 32 + kb + 16];
    v16bf af = __builtin_shufflevector(aLo, aHi, 0, 1, 2, 3, 4, 5, 6, 7,
                                       8, 9, 10, 11, 12, 13, 14, 15);
    // ---- B fragments (transposed): two contiguous 16B runs per lane, same k order as A ----
    const int kb2  = (lane < 16) ? 0 : 8;
    const int colA = (lane & 15) + (wave * 2 + 0) * 16;
    const int colB = (lane & 15) + (wave * 2 + 1) * 16;
    v8bf b0a = *(const v8bf*)&lBt[colA * 32 + kb2];
    v8bf b0b = *(const v8bf*)&lBt[colA * 32 + kb2 + 16];
    v16bf bfA = __builtin_shufflevector(b0a, b0b, 0, 1, 2, 3, 4, 5, 6, 7,
                                        8, 9, 10, 11, 12, 13, 14, 15);
    v8bf b1a = *(const v8bf*)&lBt[colB * 32 + kb2];
    v8bf b1b = *(const v8bf*)&lBt[colB * 32 + kb2 + 16];
    v16bf bfB = __builtin_shufflevector(b1a, b1b, 0, 1, 2, 3, 4, 5, 6, 7,
                                        8, 9, 10, 11, 12, 13, 14, 15);

    acc0 = __builtin_amdgcn_wmma_f32_16x16x32_bf16(false, af, false, bfA,
                                                   (short)0, acc0, false, false);
    acc1 = __builtin_amdgcn_wmma_f32_16x16x32_bf16(false, af, false, bfB,
                                                   (short)0, acc1, false, false);
    __syncthreads();
  }

  // ---- epilogue + store (C layout: VGPR r -> row r or 8+r; col = lane&15) ----
  const int rbase = (lane < 16) ? 0 : 8;
  #pragma unroll
  for (int t = 0; t < 2; ++t) {
    const int col = n0 + (wave * 2 + t) * 16 + (lane & 15);
    if (col >= N) continue;
    const float bv = (flags & F_BIAS) ? bias[col] : 0.f;
    #pragma unroll
    for (int r = 0; r < 8; ++r) {
      const int m = m0 + rbase + r;
      if (m < M) {
        float v = (t == 0 ? acc0[r] : acc1[r]) + bv;
        if (flags & F_PE) {
          const int l  = (flags & F_SUBJ) ? m : (m % L_);
          const int i2 = col >> 1;
          const float ang = (float)l * powf(10000.f, -2.f * (float)i2 / (float)D_);
          v += (col & 1) ? cosf(ang) : sinf(ang);
        }
        if (flags & F_GELU) {
          const float x = v;
          v = 0.5f * x * (1.f + tanhf(0.7978845608f * (x + 0.044715f * x * x * x)));
        }
        if (flags & F_RES) v += res[(long)m * ldc + col];
        Cout[(long)m * ldc + col] = v;
      }
    }
  }
}

// ---------------- layernorm: one wave per token ----------------
__global__ __launch_bounds__(256)
void k_ln(const float* __restrict__ x, float* __restrict__ y,
          const float* __restrict__ sc, const float* __restrict__ bi)
{
  const int tok  = blockIdx.x * 8 + (threadIdx.x >> 5);
  const int lane = threadIdx.x & 31;
  const float* xr = x + (long)tok * D_;
  float s = 0.f, s2 = 0.f;
  for (int j = lane; j < D_; j += 32) { const float v = xr[j]; s += v; s2 += v * v; }
  s = wave_sum(s); s2 = wave_sum(s2);
  const float mean = s * (1.f / D_);
  const float var  = s2 * (1.f / D_) - mean * mean;
  const float inv  = rsqrtf(var + LN_EPS_);
  float* yr = y + (long)tok * D_;
  for (int j = lane; j < D_; j += 32)
    yr[j] = (xr[j] - mean) * inv * sc[j] + bi[j];
}

// ---------------- attention: one wave per (head, batch), L=30 ----------------
__global__ __launch_bounds__(32)
void k_attn(const float* __restrict__ qkv, float* __restrict__ ao)
{
  const int h = blockIdx.x, b = blockIdx.y, lane = threadIdx.x;
  __shared__ float Q[L_][DH_], Ks[L_][DH_], Vs[L_][DH_];
  const float* base = qkv + (long)b * L_ * (3 * D_);
  for (int i = lane; i < L_ * DH_; i += 32) {
    const int l = i / DH_, d = i % DH_;
    Q[l][d]  = base[(long)l * 3 * D_ + h * DH_ + d];
    Ks[l][d] = base[(long)l * 3 * D_ + D_ + h * DH_ + d];
    Vs[l][d] = base[(long)l * 3 * D_ + 2 * D_ + h * DH_ + d];
  }
  __syncthreads();
  const float scale = 0.125f;  // 1/sqrt(64)
  for (int q = 0; q < L_; ++q) {
    float s = -1e30f;
    if (lane < L_) {
      s = 0.f;
      #pragma unroll
      for (int d = 0; d < DH_; ++d) s += Q[q][d] * Ks[lane][d];
      s *= scale;
    }
    const float mx = wave_max(s);
    const float e  = (lane < L_) ? __expf(s - mx) : 0.f;
    const float p  = e / wave_sum(e);
    float a0 = 0.f, a1 = 0.f;
    for (int k = 0; k < L_; ++k) {
      const float pk = __shfl(p, k, 32);
      a0 += pk * Vs[k][2 * lane];
      a1 += pk * Vs[k][2 * lane + 1];
    }
    float* o = ao + ((long)(b * L_ + q)) * D_ + h * DH_ + 2 * lane;
    o[0] = a0; o[1] = a1;
  }
}

// ---------------- codebook L2-normalize: one wave per code ----------------
__global__ __launch_bounds__(256)
void k_cbnorm(const float* __restrict__ cb, float* __restrict__ cbn)
{
  const int c    = blockIdx.x * 8 + (threadIdx.x >> 5);
  const int lane = threadIdx.x & 31;
  const float v0 = cb[(long)c * DC_ + lane];
  const float v1 = cb[(long)c * DC_ + 32 + lane];
  const float ss = wave_sum(v0 * v0 + v1 * v1);
  const float inv = 1.f / (sqrtf(ss) + 1e-6f);
  cbn[(long)c * DC_ + lane]      = v0 * inv;
  cbn[(long)c * DC_ + 32 + lane] = v1 * inv;
}

// ---------------- VQ: normalize z, argmax cosine sim, gather zq, vq-loss ----
__global__ __launch_bounds__(32)
void k_vq(const float* __restrict__ z, const float* __restrict__ cbn,
          float* __restrict__ zq, int* __restrict__ idx_out,
          float* __restrict__ acc)
{
  const int t = blockIdx.x, lane = threadIdx.x;
  __shared__ float znS[DC_];
  const float z0 = z[(long)t * DC_ + lane];
  const float z1 = z[(long)t * DC_ + 32 + lane];
  const float inv = 1.f / (sqrtf(wave_sum(z0 * z0 + z1 * z1)) + 1e-6f);
  znS[lane] = z0 * inv; znS[lane + 32] = z1 * inv;
  __syncthreads();

  // keep the normalized vector in registers for the 8192-code scan
  float zn[DC_];
  #pragma unroll
  for (int d = 0; d < DC_; ++d) zn[d] = znS[d];

  float best = -1e30f; int bi = 0;
  for (int c = lane; c < K_; c += 32) {
    const float* cp = cbn + (long)c * DC_;
    float s = 0.f;
    #pragma unroll 8
    for (int d = 0; d < DC_; ++d) s += zn[d] * cp[d];
    if (s > best) { best = s; bi = c; }
  }
  #pragma unroll
  for (int off = 16; off > 0; off >>= 1) {
    const float ob = __shfl_xor(best, off, 32);
    const int   oi = __shfl_xor(bi,   off, 32);
    if (ob > best || (ob == best && oi < bi)) { best = ob; bi = oi; }
  }
  const float* cp = cbn + (long)bi * DC_;
  const float q0 = cp[lane], q1 = cp[lane + 32];
  zq[(long)t * DC_ + lane]      = q0;   // straight-through forward value
  zq[(long)t * DC_ + 32 + lane] = q1;
  const float d0 = zn[lane] - q0, d1 = zn[lane + 32] - q1;
  const float ls = wave_sum(d0 * d0 + d1 * d1);
  if (lane == 0) { idx_out[t] = bi; atomicAdd(&acc[0], ls); }
}

// ---------------- reconstruction loss: one wave per (b, c) ----------------
__global__ __launch_bounds__(256)
void k_loss_bc(const float* __restrict__ X, const float* __restrict__ Xr,
               const float* __restrict__ mask, float* __restrict__ acc)
{
  const int w = blockIdx.x * 8 + (threadIdx.x >> 5);
  const int lane = threadIdx.x & 31;
  const int b = w / C_, c = w % C_;
  float s = 0.f;
  for (int t = lane; t < SEQ_; t += 32) {
    const long i = ((long)b * SEQ_ + t) * C_ + c;
    const float d = X[i] - Xr[i];
    s += d * d;
  }
  s = wave_sum(s);
  if (lane == 0) {
    const float m = mask[b * C_ + c];
    atomicAdd(&acc[1], (s / (float)SEQ_) * m);
    atomicAdd(&acc[2], m);
  }
}

__global__ void k_init(float* acc) { if (threadIdx.x < 16) acc[threadIdx.x] = 0.f; }

__global__ void k_final(const float* __restrict__ acc, float* __restrict__ out)
{
  const float loss_vq  = acc[0] / (float)(MTOK_ * DC_);   // BETA = 1
  const float loss_rgs = acc[1] / (acc[2] + 1e-12f);
  out[0] = loss_rgs + loss_vq;  // loss_total (RGS_SCALE = VQ_SCALE = 1)
  out[1] = loss_vq;
  out[2] = loss_rgs;
}

// =====================================================================
// host-side orchestration
// =====================================================================
extern "C" void kernel_launch(void* const* d_in, const int* in_sizes, int n_in,
                              void* d_out, int out_size, void* d_ws, size_t ws_size,
                              hipStream_t stream) {
  (void)in_sizes; (void)n_in; (void)out_size; (void)ws_size;
  // inputs in setup_inputs() insertion order (nested dicts flattened in key
  // insertion order): X, subj_idx, channel_mask, tok_W, tok_b,
  // enc{ln1_s,ln1_b,Wqkv,bqkv,Wo,bo,ln2_s,ln2_b,W1,b1,W2,b2},
  // dec{...same...}, vq{Win,bin,codebook,Wout,bout}, rgs_W, rgs_b
  const float* X     = (const float*)d_in[0];
  const int*   subj  = (const int*)  d_in[1];
  const float* cmask = (const float*)d_in[2];
  const float* tokW  = (const float*)d_in[3];
  const float* tokB  = (const float*)d_in[4];
  const float* const* P = (const float* const*)d_in;
  const int ENC = 5, DEC = 17;
  const float* Win  = (const float*)d_in[29];
  const float* bin  = (const float*)d_in[30];
  const float* cb   = (const float*)d_in[31];
  const float* Wout = (const float*)d_in[32];
  const float* bout = (const float*)d_in[33];
  const float* rgsW = (const float*)d_in[34];
  const float* rgsB = (const float*)d_in[35];

  // workspace layout (floats)
  float* ws  = (float*)d_ws;
  float* acc = ws;                       // 16
  float* h   = ws + 16;                  // 960*768
  float* y   = h   + MTOK_ * D_;         // 960*768
  float* qkv = y   + MTOK_ * D_;         // 960*2304
  float* ao  = qkv + MTOK_ * 3 * D_;     // 960*768
  float* ff  = ao  + MTOK_ * D_;         // 960*3072
  float* z   = ff  + MTOK_ * FF_;        // 960*64
  float* zq  = z   + MTOK_ * DC_;        // 960*64
  float* cbn = zq  + MTOK_ * DC_;        // 8192*64
  int*   vix = (int*)(cbn + (long)K_ * DC_);  // 960

  float* Xr    = (float*)d_out;          // 32*3000*64
  float* louts = Xr + (long)B_ * SEQ_ * C_;

  hipLaunchKernelGGL(k_init, dim3(1), dim3(32), 0, stream, acc);

  // tokenizer: per-batch subject GEMM (30x6400)@(6400x768) + bias + pos-emb
  hipLaunchKernelGGL(k_gemm, dim3((D_ + 127) / 128, 2, B_), dim3(128), 0, stream,
                     X, PC_, tokW, D_, tokB, nullptr, h, D_,
                     L_, D_, PC_, F_BIAS | F_PE | F_SUBJ,
                     subj, (long)PC_ * D_, D_);

  auto stack = [&](int base) {
    const float* ln1s = P[base + 0]; const float* ln1b = P[base + 1];
    const float* Wqkv = P[base + 2]; const float* bqkv = P[base + 3];
    const float* Wo   = P[base + 4]; const float* bo   = P[base + 5];
    const float* ln2s = P[base + 6]; const float* ln2b = P[base + 7];
    const float* W1   = P[base + 8]; const float* b1   = P[base + 9];
    const float* W2   = P[base + 10]; const float* b2  = P[base + 11];
    for (int l = 0; l < NL_; ++l) {
      hipLaunchKernelGGL(k_ln, dim3(MTOK_ / 8), dim3(256), 0, stream,
                         h, y, ln1s + l * D_, ln1b + l * D_);
      hipLaunchKernelGGL(k_gemm, dim3((3 * D_ + 127) / 128, MTOK_ / 16, 1), dim3(128), 0, stream,
                         y, D_, Wqkv + (long)l * D_ * 3 * D_, 3 * D_,
                         bqkv + l * 3 * D_, nullptr, qkv, 3 * D_,
                         MTOK_, 3 * D_, D_, F_BIAS, nullptr, 0L, 0);
      hipLaunchKernelGGL(k_attn, dim3(H_, B_), dim3(32), 0, stream, qkv, ao);
      hipLaunchKernelGGL(k_gemm, dim3((D_ + 127) / 128, MTOK_ / 16, 1), dim3(128), 0, stream,
                         ao, D_, Wo + (long)l * D_ * D_, D_,
                         bo + l * D_, h, h, D_,
                         MTOK_, D_, D_, F_BIAS | F_RES, nullptr, 0L, 0);
      hipLaunchKernelGGL(k_ln, dim3(MTOK_ / 8), dim3(256), 0, stream,
                         h, y, ln2s + l * D_, ln2b + l * D_);
      hipLaunchKernelGGL(k_gemm, dim3((FF_ + 127) / 128, MTOK_ / 16, 1), dim3(128), 0, stream,
                         y, D_, W1 + (long)l * D_ * FF_, FF_,
                         b1 + l * FF_, nullptr, ff, FF_,
                         MTOK_, FF_, D_, F_BIAS | F_GELU, nullptr, 0L, 0);
      hipLaunchKernelGGL(k_gemm, dim3((D_ + 127) / 128, MTOK_ / 16, 1), dim3(128), 0, stream,
                         ff, FF_, W2 + (long)l * FF_ * D_, D_,
                         b2 + l * D_, h, h, D_,
                         MTOK_, D_, FF_, F_BIAS | F_RES, nullptr, 0L, 0);
    }
  };

  stack(ENC);  // encoder -> h holds E

  // VQ: z = E @ Win + bin; cosine argmax vs normalized codebook; zq @ Wout + bout + pe
  hipLaunchKernelGGL(k_gemm, dim3(1, MTOK_ / 16, 1), dim3(128), 0, stream,
                     h, D_, Win, DC_, bin, nullptr, z, DC_,
                     MTOK_, DC_, D_, F_BIAS, nullptr, 0L, 0);
  hipLaunchKernelGGL(k_cbnorm, dim3(K_ / 8), dim3(256), 0, stream, cb, cbn);
  hipLaunchKernelGGL(k_vq, dim3(MTOK_), dim3(32), 0, stream, z, cbn, zq, vix, acc);
  hipLaunchKernelGGL(k_gemm, dim3((D_ + 127) / 128, MTOK_ / 16, 1), dim3(128), 0, stream,
                     zq, DC_, Wout, D_, bout, nullptr, h, D_,
                     MTOK_, D_, DC_, F_BIAS | F_PE, nullptr, 0L, 0);

  stack(DEC);  // decoder -> h holds Td

  // regression head: per-batch subject GEMM (30x768)@(768x6400) -> Xr (== d_out)
  hipLaunchKernelGGL(k_gemm, dim3((PC_ + 127) / 128, 2, B_), dim3(128), 0, stream,
                     h, D_, rgsW, PC_, rgsB, nullptr, Xr, PC_,
                     L_, PC_, D_, F_BIAS | F_SUBJ,
                     subj, (long)D_ * PC_, PC_);

  // losses
  hipLaunchKernelGGL(k_loss_bc, dim3(B_ * C_ / 8), dim3(256), 0, stream,
                     X, Xr, cmask, acc);
  hipLaunchKernelGGL(k_final, dim3(1), dim3(1), 0, stream, acc, louts);
}
